// EDiffMG_34505767256280
// MI455X (gfx1250) — matrix operand order repaired and checked by reading, not compiled, not run on hardware
//
#include <hip/hip_runtime.h>
#include <hip/hip_bf16.h>
#include <math.h>

// ---------------- problem constants ----------------
#define NATOM   20000
#define NGRAPH  64
#define HD      128
#define EBOND   50000
#define ERAD    250000
#define ETOT    (EBOND + ERAD)
#define ADIM    6
#define ZDIM    134           // ADIM + HD
#define ZPAD    160           // padded K for the inp GEMM (mult of 32, rows 16B aligned)
#define CUTG    10.0f
#define CUTL    2.1f

// ---------------- input index map (jax tree: dicts sorted by key) ----------
enum {
  IN_ATOM = 0, IN_POS = 1,
  EG_BOND = 2, EG_L1B = 3, EG_L1W = 4, EG_L2B = 5, EG_L2W = 6,
  EL_BOND = 7, EL_L1B = 8, EL_L1W = 9, EL_L2B = 10, EL_L2W = 11,
  G_BLK0 = 12,              // 3 blocks x 9: e1b,e1w,e2b,e2w,lin1,lin2b,lin2w,lin3b,lin3w
  G_INPB = 39, G_INPW = 40,
  L_BLK0 = 41,              // 2 blocks x 9
  L_INPB = 59, L_INPW = 60,
  GDG = 61,                 // gmlp: l1b,l1w,l2b,l2w,outb,outw
  GDL = 67,
  GNG = 73,
  GNL = 79,
  T0B = 85, T0W = 86, T1B = 87, T1W = 88, TPB = 89, TPW = 90,
  IN_EI = 91, IN_ETYPE = 92, IN_BATCH = 93, IN_TSTEP = 94
};

typedef __attribute__((ext_vector_type(16))) __bf16 v16bf;
typedef __attribute__((ext_vector_type(8)))  __bf16 v8bf;
typedef __attribute__((ext_vector_type(4)))  __bf16 v4bf;
typedef __attribute__((ext_vector_type(8)))  float  v8f;
typedef __attribute__((ext_vector_type(4)))  unsigned int u32x4;

#define ACT_NONE 0
#define ACT_SSP  2
#define ACT_GELU 3

#if __has_builtin(__builtin_amdgcn_sched_barrier)
#define SCHED_FENCE() __builtin_amdgcn_sched_barrier(0)
#else
#define SCHED_FENCE()
#endif

__device__ __forceinline__ float f_ssp(float x) {
  float sp = fmaxf(x, 0.f) + log1pf(expf(-fabsf(x)));
  return sp - 0.69314718055994531f;
}
__device__ __forceinline__ float f_gelu(float x) {
  const float c = 0.7978845608028654f;
  float t = tanhf(c * (x + 0.044715f * x * x * x));
  return 0.5f * x * (1.f + t);
}

// =====================================================================
// Tiled bf16 WMMA GEMM:  C = epilogue(A @ Bt^T + bias)
//   A : (M,K) bf16 row-major (rows 16B-aligned, K % 32 == 0)
//   Bt: (N,K) bf16 row-major (weights pre-transposed+padded), N % 128 == 0
// BM=128 x BN=128 x BK=32; 256 thr = 8 waves; wave owns 16 rows -> 8 wmma
// per K-step. Double-buffered LDS + register prefetch. All 16 fragment
// ds_loads are pinned ABOVE the wmma chain with a sched_barrier so they
// pipeline together (one DS latency per K-step, not eight).
// =====================================================================
#define BM 128
#define BN 128
#define BK 32
#define ASTR (BK + 8)   // 40 bf16: 80B row stride, keeps 16B alignment

__global__ __launch_bounds__(256, 2)
void gemm_bf16_wmma(const __bf16* __restrict__ A, int lda,
                    const __bf16* __restrict__ Bt, int ldb,
                    const float* __restrict__ bias,
                    float* __restrict__ Cf, __bf16* __restrict__ Cb, int ldc,
                    int M, int K, int N, int act,
                    const float* __restrict__ scaleMat,  // [*,N] via scaleIdx[row]
                    const int*   __restrict__ scaleIdx,
                    const float* __restrict__ addMat,    // [*,N] via addIdx[row]
                    const int*   __restrict__ addIdx,
                    const float* __restrict__ rowScale,  // per-row scalar
                    const float* __restrict__ residual, int ldres)
{
  __shared__ __attribute__((aligned(32))) __bf16 As[2][BM][ASTR];
  __shared__ __attribute__((aligned(32))) __bf16 Bs[2][BN][BK];

  const int tid  = threadIdx.x;
  const int lane = tid & 31;
  const int wid  = tid >> 5;
  const int bm   = blockIdx.y * BM;
  const int bn   = blockIdx.x * BN;

  // staging coords: 2 threads per row, 16 bf16 (32B) each
  const int srow = tid >> 1;
  const int scol = (tid & 1) * 16;
  const int agr  = bm + srow;
  const __bf16* aRow = A  + (size_t)agr * lda + scol;
  const __bf16* bRow = Bt + (size_t)(bn + srow) * ldb + scol;
  const bool aOk = (agr < M);

  v8f acc[8];
  #pragma unroll
  for (int n = 0; n < 8; ++n)
    #pragma unroll
    for (int i = 0; i < 8; ++i) acc[n][i] = 0.f;

  const int ktiles = K / BK;
  u32x4 a0, a1, b0, b1;

  auto loadT = [&](int k0) {
    if (aOk) {
      const u32x4* p = (const u32x4*)(aRow + k0);
      a0 = p[0]; a1 = p[1];
    } else {
      a0 = (u32x4){0, 0, 0, 0}; a1 = (u32x4){0, 0, 0, 0};
    }
    const u32x4* q = (const u32x4*)(bRow + k0);
    b0 = q[0]; b1 = q[1];
  };
  auto storeT = [&](int buf) {
    u32x4* pa = (u32x4*)&As[buf][srow][scol];
    pa[0] = a0; pa[1] = a1;
    u32x4* pb = (u32x4*)&Bs[buf][srow][scol];
    pb[0] = b0; pb[1] = b1;
  };

  loadT(0);
  storeT(0);
  __syncthreads();

  const int fr  = lane & 15;
  const int kb  = (lane < 16) ? 0 : 8;    // A: 16-bit 16x32 layout
  const int kb2 = (lane < 16) ? 0 : 8;    // B: same 16-bit 16x32 layout as A

  for (int kt = 0; kt < ktiles; ++kt) {
    const int cur = kt & 1;
    const bool more = (kt + 1 < ktiles);
    if (more) loadT((kt + 1) * BK);
    if (aOk && kt + 2 < ktiles)
      __builtin_prefetch((const void*)(aRow + (kt + 2) * BK), 0, 1);

    // A fragment for this wave's 16 rows
    const __bf16* ap = &As[cur][wid * 16 + fr][kb];
    v8bf alo = *(const v8bf*)(ap);
    v8bf ahi = *(const v8bf*)(ap + 16);
    v16bf af = __builtin_shufflevector(alo, ahi,
                 0, 1, 2, 3, 4, 5, 6, 7, 8, 9, 10, 11, 12, 13, 14, 15);

    // Preload ALL 8 B fragments, then fence the scheduler so the loads
    // cannot sink to their uses: 16 ds_load_b128 pipeline together, the
    // 8-wmma chain then issues back-to-back.
    v16bf barr[8];
    #pragma unroll
    for (int n = 0; n < 8; ++n)
      barr[n] = __builtin_shufflevector(*(const v8bf*)&Bs[cur][n * 16 + fr][kb2], *(const v8bf*)&Bs[cur][n * 16 + fr][kb2 + 16], 0, 1, 2, 3, 4, 5, 6, 7, 8, 9, 10, 11, 12, 13, 14, 15);
    SCHED_FENCE();
    #pragma unroll
    for (int n = 0; n < 8; ++n)
      acc[n] = __builtin_amdgcn_wmma_f32_16x16x32_bf16(
          false, af, false, barr[n], (short)0, acc[n], false, false);
    SCHED_FENCE();

    if (more) storeT(cur ^ 1);
    __syncthreads();
  }

  // epilogue: elem i -> m = i + (lane<16?0:8); n = (lane&15) + 16*nsub
  const int nlane = lane & 15;
  const int mo    = (lane < 16) ? 0 : 8;
  #pragma unroll
  for (int nsub = 0; nsub < 8; ++nsub) {
    int col = bn + nsub * 16 + nlane;
    if (col >= N) continue;
    float bv = bias ? bias[col] : 0.f;
    #pragma unroll
    for (int i = 0; i < 8; ++i) {
      int row = bm + wid * 16 + mo + i;
      if (row >= M) continue;
      float v = acc[nsub][i] + bv;
      if      (act == ACT_SSP)  v = f_ssp(v);
      else if (act == ACT_GELU) v = f_gelu(v);
      if (rowScale) v *= rowScale[row];
      if (scaleMat) v *= scaleMat[(size_t)scaleIdx[row] * N + col];
      if (addMat)   v += addMat[(size_t)addIdx[row] * N + col];
      if (residual) v += residual[(size_t)row * ldres + col];
      if (Cf) Cf[(size_t)row * ldc + col] = v;
      if (Cb) Cb[(size_t)row * ldc + col] = (__bf16)v;
    }
  }
}

// =====================================================================
// Helper kernels
// =====================================================================

// weight convert + transpose + K-pad:  out(N,Kpad) <- w(K,N)
__global__ void wconvt_kernel(const float* __restrict__ w, int K, int N, int Kpad,
                              __bf16* __restrict__ out)
{
  long long i = (long long)blockIdx.x * blockDim.x + threadIdx.x;
  if (i >= (long long)N * Kpad) return;
  int n = (int)(i / Kpad), k = (int)(i % Kpad);
  out[i] = (k < K) ? (__bf16)w[(size_t)k * N + n] : (__bf16)0.f;
}

// f32 -> bf16
__global__ void cvt_bf16_kernel(const float* __restrict__ in, __bf16* __restrict__ out,
                                long long n)
{
  long long i = (long long)blockIdx.x * blockDim.x + threadIdx.x;
  if (i < n) out[i] = (__bf16)in[i];
}

// timestep embedding + temb MLP (per-graph block, 128 threads)
__global__ void temb_kernel(const int* __restrict__ ts,
                            const float* __restrict__ w0, const float* __restrict__ b0,
                            const float* __restrict__ w1, const float* __restrict__ b1,
                            const float* __restrict__ wp, const float* __restrict__ bp,
                            float* __restrict__ temb)
{
  __shared__ float emb[HD];
  __shared__ float a[4 * HD];
  __shared__ float c[4 * HD];
  int g = blockIdx.x, tid = threadIdx.x;
  float tv = (float)ts[g];
  if (tid < 64) {
    float f = expf(-9.210340371976184f * (float)tid / 63.f);
    float arg = tv * f;
    emb[tid]      = sinf(arg);
    emb[tid + 64] = cosf(arg);
  }
  __syncthreads();
  for (int o = tid; o < 512; o += 128) {
    float s = b0[o];
    for (int k = 0; k < 128; ++k) s += emb[k] * w0[k * 512 + o];
    a[o] = fmaxf(s, 0.f);
  }
  __syncthreads();
  for (int o = tid; o < 512; o += 128) {
    float s = b1[o];
    for (int k = 0; k < 512; ++k) s += a[k] * w1[k * 512 + o];
    c[o] = fmaxf(s, 0.f);   // relu before temb_proj
  }
  __syncthreads();
  {
    int o = tid;
    float s = bp[o];
    for (int k = 0; k < 512; ++k) s += c[k] * wp[k * 128 + o];
    temb[(size_t)g * HD + o] = s;
  }
}

// per-edge distance, cutoffs, temb gather index
__global__ void edge_prep_kernel(const float* __restrict__ pos,
                                 const int* __restrict__ src, const int* __restrict__ dst,
                                 const int* __restrict__ batch,
                                 float* __restrict__ d, float* __restrict__ cg,
                                 float* __restrict__ cl, int* __restrict__ tidx, int E)
{
  int e = blockIdx.x * blockDim.x + threadIdx.x;
  if (e >= E) return;
  int s = src[e], t = dst[e];
  float dx = pos[s * 3 + 0] - pos[t * 3 + 0];
  float dy = pos[s * 3 + 1] - pos[t * 3 + 1];
  float dz = pos[s * 3 + 2] - pos[t * 3 + 2];
  float dd = sqrtf(dx * dx + dy * dy + dz * dz);
  d[e] = dd;
  cg[e] = (dd <= CUTG) ? 0.5f * (cosf(dd * (3.14159265358979323846f / CUTG)) + 1.f) : 0.f;
  cl[e] = (dd <= CUTL) ? 0.5f * (cosf(dd * (3.14159265358979323846f / CUTL)) + 1.f) : 0.f;
  tidx[e] = batch[s];
}

// 1->H linear + relu, bf16 out (edge encoder l1)
__global__ void edge_l1_kernel(const float* __restrict__ d,
                               const float* __restrict__ w, const float* __restrict__ b,
                               __bf16* __restrict__ out, int E)
{
  long long i = (long long)blockIdx.x * blockDim.x + threadIdx.x;
  if (i >= (long long)E * HD) return;
  int e = (int)(i >> 7), j = (int)(i & (HD - 1));
  out[i] = (__bf16)fmaxf(d[e] * w[j] + b[j], 0.f);
}

// z = concat(atom_type, temb[batch]) -> bf16 with ZPAD stride (zero padded)
__global__ void zcat_kernel(const float* __restrict__ atom, const float* __restrict__ temb,
                            const int* __restrict__ batch, __bf16* __restrict__ z)
{
  long long i = (long long)blockIdx.x * blockDim.x + threadIdx.x;
  if (i >= (long long)NATOM * ZPAD) return;
  int n = (int)(i / ZPAD), j = (int)(i % ZPAD);
  float v = 0.f;
  if (j < ADIM)      v = atom[(size_t)n * ADIM + j];
  else if (j < ZDIM) v = temb[(size_t)batch[n] * HD + (j - ADIM)];
  z[i] = (__bf16)v;
}

__global__ void zero_kernel(float* __restrict__ p, long long n) {
  long long i = (long long)blockIdx.x * blockDim.x + threadIdx.x;
  if (i < n) p[i] = 0.f;
}

// segment_sum, one wave per edge: agg[dst] += x[src] * W   (float4 per lane)
__global__ void scatter_edge_kernel(const float* __restrict__ x, const float* __restrict__ W,
                                    const int* __restrict__ src, const int* __restrict__ dst,
                                    float* __restrict__ agg, int E)
{
  int lane = threadIdx.x & 31;
  int e = blockIdx.x * 8 + (threadIdx.x >> 5);
  if (e >= E) return;
  int s = src[e], t = dst[e];
  float4 xv = ((const float4*)(x + (size_t)s * HD))[lane];
  float4 wv = ((const float4*)(W + (size_t)e * HD))[lane];
  float* ap = agg + (size_t)t * HD + lane * 4;
  atomicAdd(ap + 0, xv.x * wv.x);
  atomicAdd(ap + 1, xv.y * wv.y);
  atomicAdd(ap + 2, xv.z * wv.z);
  atomicAdd(ap + 3, xv.w * wv.w);
}

// hp = concat(h[src]*h[dst], ea) -> bf16 (E x 256), one wave per edge
__global__ void hp_edge_kernel(const float* __restrict__ h, const __bf16* __restrict__ ea,
                               const int* __restrict__ src, const int* __restrict__ dst,
                               __bf16* __restrict__ hp, int E)
{
  int lane = threadIdx.x & 31;
  int e = blockIdx.x * 8 + (threadIdx.x >> 5);
  if (e >= E) return;
  int s = src[e], t = dst[e];
  float4 a = ((const float4*)(h + (size_t)s * HD))[lane];
  float4 b = ((const float4*)(h + (size_t)t * HD))[lane];
  v4bf pv = { (__bf16)(a.x * b.x), (__bf16)(a.y * b.y),
              (__bf16)(a.z * b.z), (__bf16)(a.w * b.w) };
  *(v4bf*)(hp + (size_t)e * 256 + lane * 4) = pv;
  const unsigned long long* ep = (const unsigned long long*)(ea + (size_t)e * HD);
  *(unsigned long long*)(hp + (size_t)e * 256 + HD + lane * 4) = ep[lane];
}

// output head, one wave per row, DOUT compile-time (1 or 6)
template <int DOUT>
__global__ void waverow_out_kernel(const __bf16* __restrict__ A, int K,
                                   const float* __restrict__ W, const float* __restrict__ b,
                                   float* __restrict__ out, long long M)
{
  int lane = threadIdx.x & 31;
  long long row = (long long)blockIdx.x * 8 + (threadIdx.x >> 5);
  if (row >= M) return;
  const __bf16* a = A + (size_t)row * K;
  float s[DOUT];
  #pragma unroll
  for (int j = 0; j < DOUT; ++j) s[j] = 0.f;
  for (int k = lane; k < K; k += 32) {
    float av = (float)a[k];
    #pragma unroll
    for (int j = 0; j < DOUT; ++j) s[j] += av * W[k * DOUT + j];
  }
  #pragma unroll
  for (int j = 0; j < DOUT; ++j) {
    float v = s[j];
    #pragma unroll
    for (int off = 16; off > 0; off >>= 1) v += __shfl_xor(v, off, 32);
    if (lane == 0) out[row * DOUT + j] = v + b[j];
  }
}

// =====================================================================
// Host helpers
// =====================================================================
static inline unsigned blocks_for(long long n) { return (unsigned)((n + 255) / 256); }
static inline unsigned wblocks_for(long long n) { return (unsigned)((n + 7) / 8); }

static void run_gemm(hipStream_t st, const __bf16* A, int lda, const __bf16* Bt, int ldb,
                     const float* bias, float* Cf, __bf16* Cb, int ldc,
                     int M, int K, int N, int act,
                     const float* sm = nullptr, const int* si = nullptr,
                     const float* am = nullptr, const int* ai = nullptr,
                     const float* rs = nullptr, const float* res = nullptr, int ldres = 0)
{
  dim3 grid((N + BN - 1) / BN, (M + BM - 1) / BM);
  gemm_bf16_wmma<<<grid, 256, 0, st>>>(A, lda, Bt, ldb, bias, Cf, Cb, ldc, M, K, N, act,
                                       sm, si, am, ai, rs, res, ldres);
}

struct BlkW { const __bf16 *e1, *e2, *l1, *l2, *l3; };

static void run_block(hipStream_t st, void* const* din, int pb, const BlkW& w,
                      const __bf16* ea, const float* cut,
                      const int* src, const int* dst, int Eloc,
                      float* h, __bf16* h_bf,
                      __bf16* te_bf, float* Wbuf, float* xb,
                      float* agg, __bf16* agg_bf, __bf16* t2_bf)
{
  const float* e1b = (const float*)din[pb + 0];
  const float* e2b = (const float*)din[pb + 2];
  const float* l2b = (const float*)din[pb + 5];
  const float* l3b = (const float*)din[pb + 7];

  run_gemm(st, ea, HD, w.e1, HD, e1b, nullptr, te_bf, HD, Eloc, HD, HD, ACT_SSP);
  run_gemm(st, te_bf, HD, w.e2, HD, e2b, Wbuf, nullptr, HD, Eloc, HD, HD, ACT_NONE,
           nullptr, nullptr, nullptr, nullptr, cut, nullptr, 0);
  run_gemm(st, h_bf, HD, w.l1, HD, nullptr, xb, nullptr, HD, NATOM, HD, HD, ACT_NONE);
  zero_kernel<<<blocks_for((long long)NATOM * HD), 256, 0, st>>>(agg, (long long)NATOM * HD);
  scatter_edge_kernel<<<wblocks_for(Eloc), 256, 0, st>>>(xb, Wbuf, src, dst, agg, Eloc);
  cvt_bf16_kernel<<<blocks_for((long long)NATOM * HD), 256, 0, st>>>(agg, agg_bf,
                                                                     (long long)NATOM * HD);
  run_gemm(st, agg_bf, HD, w.l2, HD, l2b, nullptr, t2_bf, HD, NATOM, HD, HD, ACT_SSP);
  run_gemm(st, t2_bf, HD, w.l3, HD, l3b, h, h_bf, HD, NATOM, HD, HD, ACT_NONE,
           nullptr, nullptr, nullptr, nullptr, nullptr, h, HD);
}

template <int DOUT>
static void run_gmlp(hipStream_t st, void* const* din, int pb,
                     const __bf16* wl1, const __bf16* wl2,
                     const __bf16* X, int dinK, long long M,
                     __bf16* tmp1, __bf16* tmp2, float* out)
{
  const float* l1b = (const float*)din[pb + 0];
  const float* l2b = (const float*)din[pb + 2];
  const float* ob  = (const float*)din[pb + 4];
  const float* ow  = (const float*)din[pb + 5];
  run_gemm(st, X, dinK, wl1, dinK, l1b, nullptr, tmp1, dinK, (int)M, dinK, dinK, ACT_GELU);
  run_gemm(st, tmp1, dinK, wl2, dinK, l2b, nullptr, tmp2, dinK, (int)M, dinK, dinK, ACT_GELU);
  waverow_out_kernel<DOUT><<<wblocks_for(M), 256, 0, st>>>(tmp2, dinK, ow, ob, out, M);
}

// =====================================================================
extern "C" void kernel_launch(void* const* d_in, const int* in_sizes, int n_in,
                              void* d_out, int out_size, void* d_ws, size_t ws_size,
                              hipStream_t stream)
{
  (void)in_sizes; (void)n_in; (void)out_size; (void)ws_size;

  const float* atom  = (const float*)d_in[IN_ATOM];
  const float* pos   = (const float*)d_in[IN_POS];
  const int*   ei    = (const int*)d_in[IN_EI];
  const int*   etype = (const int*)d_in[IN_ETYPE];
  const int*   batch = (const int*)d_in[IN_BATCH];
  const int*   tstep = (const int*)d_in[IN_TSTEP];
  float* out = (float*)d_out;

  char* ws = (char*)d_ws;
  size_t off = 0;
  auto alloc = [&](size_t bytes) -> void* {
    void* p = ws + off;
    off = (off + bytes + 255) & ~(size_t)255;
    return p;
  };
  float*  temb   = (float*) alloc((size_t)NGRAPH * HD * 4);
  float*  dbuf   = (float*) alloc((size_t)ETOT * 4);
  float*  cutg   = (float*) alloc((size_t)ETOT * 4);
  float*  cutl   = (float*) alloc((size_t)ETOT * 4);
  int*    tidx   = (int*)   alloc((size_t)ETOT * 4);
  __bf16* z_bf   = (__bf16*)alloc((size_t)NATOM * ZPAD * 2);
  float*  hg     = (float*) alloc((size_t)NATOM * HD * 4);
  float*  hl     = (float*) alloc((size_t)NATOM * HD * 4);
  __bf16* hg_bf  = (__bf16*)alloc((size_t)NATOM * HD * 2);
  __bf16* hl_bf  = (__bf16*)alloc((size_t)NATOM * HD * 2);
  float*  xb     = (float*) alloc((size_t)NATOM * HD * 4);
  float*  agg    = (float*) alloc((size_t)NATOM * HD * 4);
  __bf16* agg_bf = (__bf16*)alloc((size_t)NATOM * HD * 2);
  __bf16* t2_bf  = (__bf16*)alloc((size_t)NATOM * HD * 2);
  __bf16* nb1_bf = (__bf16*)alloc((size_t)NATOM * HD * 2);
  __bf16* nb2_bf = (__bf16*)alloc((size_t)NATOM * HD * 2);
  __bf16* te_bf  = (__bf16*)alloc((size_t)ETOT * HD * 2);
  __bf16* ea_bf  = (__bf16*)alloc((size_t)ETOT * HD * 2);
  float*  Wbuf   = (float*) alloc((size_t)ETOT * HD * 4);
  __bf16* hp_bf  = (__bf16*)alloc((size_t)ETOT * 256 * 2);
  __bf16* g1_bf  = (__bf16*)alloc((size_t)ETOT * 256 * 2);
  __bf16* arena  = (__bf16*)alloc((size_t)2 * 1024 * 1024);

  // ---- weight prep: convert + transpose + pad to bf16 (N, Kpad) ----
  size_t aoff = 0;
  auto wt = [&](int idx, int K, int N, int Kpad) -> const __bf16* {
    __bf16* p = arena + aoff;
    aoff += (size_t)N * Kpad;
    wconvt_kernel<<<blocks_for((long long)N * Kpad), 256, 0, stream>>>(
        (const float*)d_in[idx], K, N, Kpad, p);
    return p;
  };
  const __bf16* w_eg_l2 = wt(EG_L2W, HD, HD, HD);
  const __bf16* w_el_l2 = wt(EL_L2W, HD, HD, HD);
  const __bf16* w_ginp  = wt(G_INPW, ZDIM, HD, ZPAD);
  const __bf16* w_linp  = wt(L_INPW, ZDIM, HD, ZPAD);
  BlkW gblk[3], lblk[2];
  for (int b = 0; b < 3; ++b) {
    int pb = G_BLK0 + 9 * b;
    gblk[b] = { wt(pb + 1, HD, HD, HD), wt(pb + 3, HD, HD, HD), wt(pb + 4, HD, HD, HD),
                wt(pb + 6, HD, HD, HD), wt(pb + 8, HD, HD, HD) };
  }
  for (int b = 0; b < 2; ++b) {
    int pb = L_BLK0 + 9 * b;
    lblk[b] = { wt(pb + 1, HD, HD, HD), wt(pb + 3, HD, HD, HD), wt(pb + 4, HD, HD, HD),
                wt(pb + 6, HD, HD, HD), wt(pb + 8, HD, HD, HD) };
  }
  const __bf16* w_gd1 = wt(GDG + 1, 256, 256, 256);
  const __bf16* w_gd2 = wt(GDG + 3, 256, 256, 256);
  const __bf16* w_ld1 = wt(GDL + 1, 256, 256, 256);
  const __bf16* w_ld2 = wt(GDL + 3, 256, 256, 256);
  const __bf16* w_gn1 = wt(GNG + 1, HD, HD, HD);
  const __bf16* w_gn2 = wt(GNG + 3, HD, HD, HD);
  const __bf16* w_ln1 = wt(GNL + 1, HD, HD, HD);
  const __bf16* w_ln2 = wt(GNL + 3, HD, HD, HD);

  const int* srcG = ei;
  const int* dstG = ei + ETOT;
  const int* srcL = ei + EBOND;
  const int* dstL = ei + ETOT + EBOND;

  float* out_dist_g = out;
  float* out_dist_l = out + ETOT;
  float* out_node_g = out + ETOT + ERAD;
  float* out_node_l = out + ETOT + ERAD + (size_t)NATOM * ADIM;

  // ---- small front-end ----
  temb_kernel<<<NGRAPH, 128, 0, stream>>>(tstep,
      (const float*)d_in[T0W], (const float*)d_in[T0B],
      (const float*)d_in[T1W], (const float*)d_in[T1B],
      (const float*)d_in[TPW], (const float*)d_in[TPB], temb);
  edge_prep_kernel<<<blocks_for(ETOT), 256, 0, stream>>>(pos, srcG, dstG, batch,
                                                         dbuf, cutg, cutl, tidx, ETOT);
  zcat_kernel<<<blocks_for((long long)NATOM * ZPAD), 256, 0, stream>>>(atom, temb, batch, z_bf);

  // ============== GLOBAL branch ==============
  edge_l1_kernel<<<blocks_for((long long)ETOT * HD), 256, 0, stream>>>(
      dbuf, (const float*)d_in[EG_L1W], (const float*)d_in[EG_L1B], te_bf, ETOT);
  run_gemm(stream, te_bf, HD, w_eg_l2, HD, (const float*)d_in[EG_L2B],
           nullptr, ea_bf, HD, ETOT, HD, HD, ACT_NONE,
           (const float*)d_in[EG_BOND], etype, temb, tidx, nullptr, nullptr, 0);

  run_gemm(stream, z_bf, ZPAD, w_ginp, ZPAD, (const float*)d_in[G_INPB],
           hg, hg_bf, HD, NATOM, ZPAD, HD, ACT_NONE);
  for (int b = 0; b < 3; ++b)
    run_block(stream, d_in, G_BLK0 + 9 * b, gblk[b], ea_bf, cutg, srcG, dstG, ETOT,
              hg, hg_bf, te_bf, Wbuf, xb, agg, agg_bf, t2_bf);

  hp_edge_kernel<<<wblocks_for(ETOT), 256, 0, stream>>>(hg, ea_bf, srcG, dstG, hp_bf, ETOT);
  run_gmlp<1>(stream, d_in, GDG, w_gd1, w_gd2, hp_bf, 256, ETOT, g1_bf, hp_bf, out_dist_g);
  run_gmlp<ADIM>(stream, d_in, GNG, w_gn1, w_gn2, hg_bf, HD, NATOM, nb1_bf, nb2_bf,
                 out_node_g);

  // ============== LOCAL branch ==============
  edge_l1_kernel<<<blocks_for((long long)ETOT * HD), 256, 0, stream>>>(
      dbuf, (const float*)d_in[EL_L1W], (const float*)d_in[EL_L1B], te_bf, ETOT);
  run_gemm(stream, te_bf, HD, w_el_l2, HD, (const float*)d_in[EL_L2B],
           nullptr, ea_bf, HD, ETOT, HD, HD, ACT_NONE,
           (const float*)d_in[EL_BOND], etype, temb, tidx, nullptr, nullptr, 0);

  run_gemm(stream, z_bf, ZPAD, w_linp, ZPAD, (const float*)d_in[L_INPB],
           hl, hl_bf, HD, NATOM, ZPAD, HD, ACT_NONE);
  const __bf16* eaL = ea_bf + (size_t)EBOND * HD;
  for (int b = 0; b < 2; ++b)
    run_block(stream, d_in, L_BLK0 + 9 * b, lblk[b], eaL, cutl + EBOND, srcL, dstL, ERAD,
              hl, hl_bf, te_bf, Wbuf, xb, agg, agg_bf, t2_bf);

  hp_edge_kernel<<<wblocks_for(ERAD), 256, 0, stream>>>(hl, eaL, srcL, dstL, hp_bf, ERAD);
  run_gmlp<1>(stream, d_in, GDL, w_ld1, w_ld2, hp_bf, 256, ERAD, g1_bf, hp_bf, out_dist_l);
  run_gmlp<ADIM>(stream, d_in, GNL, w_ln1, w_ln2, hl_bf, HD, NATOM, nb1_bf, nb2_bf,
                 out_node_l);
}
